// GPT_49555332661284
// MI455X (gfx1250) — matrix operand order repaired and checked by reading, not compiled, not run on hardware
//
#include <hip/hip_runtime.h>
#include <hip/hip_bf16.h>

typedef _Float16 f16;
typedef __attribute__((ext_vector_type(16))) _Float16 v16h;
typedef __attribute__((ext_vector_type(8)))  _Float16 v8h;
typedef __attribute__((ext_vector_type(8)))  float    v8f;

constexpr int BB = 2, TT = 1024, DD = 1024, HH = 16, HDD = 64, LL = 4, VV = 32000;

__device__ __forceinline__ v16h cat16(v8h lo, v8h hi) {
  return __builtin_shufflevector(lo, hi, 0,1,2,3,4,5,6,7,8,9,10,11,12,13,14,15);
}
__device__ __forceinline__ v8f wmma32(v16h a, v16h b, v8f c) {
  return __builtin_amdgcn_wmma_f32_16x16x32_f16(false, a, false, b, (short)0, c, false, false);
}

// ---- gfx1250 async global->LDS DMA (ASYNCcnt), with portable fallback ----
#if defined(__AMDGCN__) && __has_builtin(__builtin_amdgcn_global_load_async_to_lds_b128)
#define HAS_ASYNC_LDS 1
#else
#define HAS_ASYNC_LDS 0
#endif
// ---- gfx1250 Tensor Data Mover (TENSORcnt) ----
#if defined(__AMDGCN__) && __has_builtin(__builtin_amdgcn_tensor_load_to_lds)
#define HAS_TDM 1
#else
#define HAS_TDM 0
#endif

typedef int i128v __attribute__((vector_size(16)));            // int4-style vector
typedef __attribute__((address_space(1))) i128v g_i128;        // global (AS1)
typedef __attribute__((address_space(3))) i128v l_i128;        // LDS (AS3)
typedef unsigned int u32x4 __attribute__((vector_size(16)));
typedef int i32x4 __attribute__((vector_size(16)));
typedef int i32x8 __attribute__((vector_size(32)));

__device__ __forceinline__ void async_b128(const f16* g, f16* l) {
#if HAS_ASYNC_LDS
  __builtin_amdgcn_global_load_async_to_lds_b128((g_i128*)g, (l_i128*)l, 0, 0);
#else
  *(uint4*)l = *(const uint4*)g;
#endif
}
__device__ __forceinline__ void stage32B(const f16* g, f16* l) {  // 32 bytes
  async_b128(g, l); async_b128(g + 8, l + 8);
}
__device__ __forceinline__ void wait_async() {
#if defined(__AMDGCN__) && __has_builtin(__builtin_amdgcn_s_wait_asynccnt)
  __builtin_amdgcn_s_wait_asynccnt(0);
#elif HAS_ASYNC_LDS
  asm volatile("s_wait_asynccnt 0x0" ::: "memory");
#endif
}

// TDM 2D tile load: tile_h rows of tile_w fp16 elems, row stride rs elems.
// tens_h < tile_h -> hardware zero-fill of OOB rows. pad_amt_code < 0 -> no LDS padding.
__device__ __forceinline__ void tdm_load_2d(const f16* gsrc, f16* ldst,
                                            int tile_w, int tile_h, int tens_h, int rs,
                                            int pad_int_code, int pad_amt_code) {
#if HAS_TDM
  unsigned long long ga = (unsigned long long)(size_t)gsrc;
  unsigned la = (unsigned)(size_t)(l_i128*)ldst;                 // LDS byte offset
  u32x4 g0 = { 1u,                                               // count=1, user D#
               la,                                               // lds_addr
               (unsigned)ga,                                     // global_addr[31:0]
               (unsigned)(ga >> 32) | (2u << 30) };              // global_addr[56:32] | type=2
  unsigned w0 = 0x00010000u;                                     // data_size = 2 bytes
  if (pad_amt_code >= 0)
    w0 |= (1u << 20) | ((unsigned)pad_int_code << 22) | ((unsigned)pad_amt_code << 25);
  i32x8 g1 = { (int)w0,
               (int)(((unsigned)tile_w & 0xFFFFu) << 16),                          // tensor_dim0 lo
               (int)(((unsigned)tile_w >> 16) | (((unsigned)tens_h & 0xFFFFu) << 16)), // dim0 hi | dim1 lo
               (int)(((unsigned)tens_h >> 16) | ((unsigned)tile_w << 16)),         // dim1 hi | tile_dim0
               (int)(unsigned)tile_h,                                              // tile_dim1 | tile_dim2=0
               (int)rs,                                                            // tensor_dim0_stride lo32
               0, 0 };
  i32x4 z4 = {};
#if __clang_major__ >= 23
  i32x8 z8 = {};
  __builtin_amdgcn_tensor_load_to_lds(g0, g1, z4, z4, z8, 0);
#else
  __builtin_amdgcn_tensor_load_to_lds(g0, g1, z4, z4, 0);
#endif
#endif
}
__device__ __forceinline__ void wait_tensor() {
#if defined(__AMDGCN__) && __has_builtin(__builtin_amdgcn_s_wait_tensorcnt)
  __builtin_amdgcn_s_wait_tensorcnt(0);
#elif HAS_TDM
  asm volatile("s_wait_tensorcnt 0x0" ::: "memory");
#endif
}

// ---------------- embedding: x = tok_emb[idx] + pos_emb ----------------
__global__ void embed_k(const int* __restrict__ idx, const float* __restrict__ tok,
                        const float* __restrict__ pos, float* __restrict__ x) {
  int i = blockIdx.x * 256 + threadIdx.x;           // i < B*T*D
  int d = i % DD, row = i / DD, t = row % TT;
  x[i] = tok[(size_t)idx[row] * DD + d] + pos[t * DD + d];
}

// ------- fused convert + transpose: W[K,N] fp32 -> Wt[N,K] fp16 (tiled) -------
__global__ __launch_bounds__(256) void cvtT_k(const float* __restrict__ s, f16* __restrict__ d,
                                              int K, int N) {
  __shared__ float t[32][33];
  const int k0 = blockIdx.y * 32, n0 = blockIdx.x * 32;
  const int tx = threadIdx.x, ty = threadIdx.y;     // 32 x 8
  #pragma unroll
  for (int j = 0; j < 32; j += 8) t[ty + j][tx] = s[(size_t)(k0 + ty + j) * N + n0 + tx];
  __syncthreads();
  #pragma unroll
  for (int j = 0; j < 32; j += 8) d[(size_t)(n0 + ty + j) * K + k0 + tx] = (f16)t[tx][ty + j];
}

// ---------------- LayerNorm (fp32 in, fp16 out) ----------------
__global__ __launch_bounds__(256) void ln_k(const float* __restrict__ x,
                                            const float* __restrict__ w,
                                            const float* __restrict__ b,
                                            f16* __restrict__ out) {
  __shared__ float s1[256], s2[256];
  const int row = blockIdx.x;
  const float* xr = x + (size_t)row * DD;
  float a = 0.f, q = 0.f;
  for (int i = threadIdx.x; i < DD; i += 256) { float v = xr[i]; a += v; q += v * v; }
  s1[threadIdx.x] = a; s2[threadIdx.x] = q; __syncthreads();
  for (int st = 128; st > 0; st >>= 1) {
    if (threadIdx.x < st) { s1[threadIdx.x] += s1[threadIdx.x + st]; s2[threadIdx.x] += s2[threadIdx.x + st]; }
    __syncthreads();
  }
  const float m  = s1[0] * (1.f / DD);
  const float rs = rsqrtf(s2[0] * (1.f / DD) - m * m + 1e-5f);
  for (int i = threadIdx.x; i < DD; i += 256)
    out[(size_t)row * DD + i] = (f16)((xr[i] - m) * rs * w[i] + b[i]);
}

// ------- WMMA GEMM: C[M,N] = A[M,K](f16) * Bt[N,K](f16, pre-transposed) + bias -------
// A tile: per-lane async DMA (ASYNCcnt, vector-memory path)
// B tile: TDM descriptor from wave 0 (TENSORcnt, DMA engine) -> both overlap with WMMA
// EPI: 0 = bias -> f16 out ; 1 = bias + residual add (fp32 in/out)
//      2 = bias + exact GELU -> f16 out ; 3 = no bias -> fp32 out (logits)
template<int EPI>
__global__ __launch_bounds__(256) void gemm_k(const f16* __restrict__ A, const f16* __restrict__ Bt,
                                              const float* __restrict__ bias,
                                              float* __restrict__ resid, f16* __restrict__ out16,
                                              float* __restrict__ out32, int M, int N, int K) {
  __shared__ f16 As[2][128 * 32];   // [M][K] tiles, ping-pong
  __shared__ f16 Bs[2][128 * 32];   // [N][K] tiles, ping-pong
  const int tid  = threadIdx.x;
  const int wave = tid >> 5, lane = tid & 31;
  const int lm = lane & 15, lh = lane >> 4;
  const int waveM = wave & 3, waveN = wave >> 2;          // 4 x 2 wave grid
  const int m0 = blockIdx.y * 128, n0 = blockIdx.x * 128;

  const int lrow = tid >> 1, lpart = (tid & 1) * 16;      // per-lane loader mapping
  const f16* ga = A  + (size_t)(m0 + lrow) * K + lpart;
  const f16* gb = Bt + (size_t)(n0 + lrow) * K + lpart;
  f16* la = (f16*)As[0] + lrow * 32 + lpart;
  f16* lb = (f16*)Bs[0] + lrow * 32 + lpart;
  constexpr int BUF = 128 * 32;
  (void)gb; (void)lb;

  v8f acc[2][4] = {};

  // prologue: stage tile 0 on both engines
#if HAS_TDM
  if (wave == 0) tdm_load_2d(Bt + (size_t)n0 * K, Bs[0], 32, 128, 128, K, 0, -1);
  stage32B(ga, la);
  wait_async();
  if (wave == 0) wait_tensor();
#else
  stage32B(ga, la);
  stage32B(gb, lb);
  wait_async();
#endif
  __syncthreads();

  int cur = 0;
  for (int k0 = 0; k0 < K; k0 += 32) {
    const int nxt = cur ^ 1;
    if (k0 + 32 < K) {                     // stage next tile while computing
#if HAS_TDM
      if (wave == 0) tdm_load_2d(Bt + (size_t)n0 * K + k0 + 32, Bs[nxt], 32, 128, 128, K, 0, -1);
      stage32B(ga + k0 + 32, la + nxt * BUF);
#else
      stage32B(ga + k0 + 32, la + nxt * BUF);
      stage32B(gb + k0 + 32, lb + nxt * BUF);
#endif
      __builtin_prefetch(ga + k0 + 64, 0, 3);   // global_prefetch_b8, 2 tiles ahead
      __builtin_prefetch(gb + k0 + 64, 0, 3);
    }

    v16h af[2], bf[4];
    #pragma unroll
    for (int wm = 0; wm < 2; wm++) {   // A frag: lane m=lm, halves K = lh*16 + j
      const f16* p = As[cur] + (waveM * 32 + wm * 16 + lm) * 32;
      af[wm] = cat16(*(const v8h*)(p + lh * 16), *(const v8h*)(p + lh * 16 + 8));
    }
    #pragma unroll
    for (int wn = 0; wn < 4; wn++) {   // B frag: lane n=lm, halves K = lh*16 + j
      const f16* p = Bs[cur] + (waveN * 64 + wn * 16 + lm) * 32 + lh * 16;
      bf[wn] = cat16(*(const v8h*)(p), *(const v8h*)(p + 8));
    }
    #pragma unroll
    for (int wm = 0; wm < 2; wm++)
      #pragma unroll
      for (int wn = 0; wn < 4; wn++)
        acc[wm][wn] = wmma32(af[wm], bf[wn], acc[wm][wn]);

    wait_async();                      // A tile landed (this wave's ASYNCcnt)
#if HAS_TDM
    if (wave == 0) wait_tensor();      // B tile landed (issuing wave's TENSORcnt)
#endif
    __syncthreads();                   // all waves done reading cur + see nxt's data
    cur = nxt;
  }

  // epilogue: C frag rows = 8*lh + r, col = lm
  #pragma unroll
  for (int wm = 0; wm < 2; wm++)
    #pragma unroll
    for (int wn = 0; wn < 4; wn++) {
      const int col = n0 + waveN * 64 + wn * 16 + lm;
      const float bv = (EPI == 3) ? 0.f : bias[col];
      #pragma unroll
      for (int r = 0; r < 8; r++) {
        const int row = m0 + waveM * 32 + wm * 16 + lh * 8 + r;
        float v = acc[wm][wn][r] + bv;
        const size_t o = (size_t)row * N + col;
        if (EPI == 0)      out16[o] = (f16)v;
        else if (EPI == 1) resid[o] += v;
        else if (EPI == 2) out16[o] = (f16)(0.5f * v * (1.f + erff(v * 0.70710678f)));
        else               out32[o] = v;
      }
    }
}

// ---------------- flash attention: one wave per 16 q-rows, 32-key chunks ----------------
__global__ __launch_bounds__(256) void attn_k(const f16* __restrict__ Q, const f16* __restrict__ Kt,
                                              const f16* __restrict__ Vt, f16* __restrict__ Y) {
  __shared__ f16 Pl[8 * 16 * 32];   // per-wave P (C-layout -> A-layout bounce)
  __shared__ f16 Vl[8 * 32 * 72];   // per-wave V chunk [32 keys][64 d], stride 72
  const int wave = threadIdx.x >> 5, lane = threadIdx.x & 31;
  const int lm = lane & 15, lh = lane >> 4;
  int bx = blockIdx.x;
  const int qb = bx & 7;  bx >>= 3;
  const int hh = bx & 15; bx >>= 4;
  const int bb = bx;
  const int q0 = qb * 128 + wave * 16;
  const size_t rowbase = (size_t)bb * TT;
  const int cb = hh * HDD;
  f16* pw = Pl + wave * 16 * 32;
  f16* vw = Vl + wave * 32 * 72;

  // Q A-frags (direct: lane m = lm, halves d = d0 + lh*16 + i)
  v16h qa[2];
  {
    const f16* qp = Q + (rowbase + q0 + lm) * DD + cb;
    #pragma unroll
    for (int g = 0; g < 2; g++)
      qa[g] = cat16(*(const v8h*)(qp + g * 32 + lh * 16), *(const v8h*)(qp + g * 32 + lh * 16 + 8));
  }

  float mi[8], li[8];
  v8f oacc[4] = {};
  #pragma unroll
  for (int r = 0; r < 8; r++) { mi[r] = -1e30f; li[r] = 0.f; }

  const int nchunks = q0 / 32 + 1;
  for (int kc = 0; kc < nchunks; kc++) {
    const int kbase = kc * 32;
    // stage V chunk [32 keys x 64 d] into this wave's LDS region
#if HAS_TDM
    {
      int remain = TT - kbase; if (remain > 32) remain = 32;   // OOB rows -> HW zero-fill
      // pad 4 dwords (code 3) every 32 dwords (code 4) => 72-half LDS row stride
      tdm_load_2d(Vt + (rowbase + kbase) * DD + cb, vw, 64, 32, remain, DD, 4, 3);
    }
#else
    {
      int kk = kbase + lane; if (kk > TT - 1) kk = TT - 1;
      const f16* vg = Vt + (rowbase + kk) * DD + cb;
      f16* vl = vw + lane * 72;
      #pragma unroll
      for (int j = 0; j < 8; j++) async_b128(vg + j * 8, vl + j * 8);
    }
#endif
    // S = Q*K^T : K^T B-frag = contiguous row of K (lane n = key lm, halves = d)
    v8f s[2];
    #pragma unroll
    for (int sn = 0; sn < 2; sn++) {
      int kk = kbase + sn * 16 + lm; int kkc = kk > TT - 1 ? TT - 1 : kk;
      const f16* kp = Kt + (rowbase + kkc) * DD + cb;
      v16h kb0 = cat16(*(const v8h*)(kp + lh * 16),      *(const v8h*)(kp + lh * 16 + 8));
      v16h kb1 = cat16(*(const v8h*)(kp + 32 + lh * 16), *(const v8h*)(kp + 32 + lh * 16 + 8));
      v8f z = {};
      z = wmma32(qa[0], kb0, z);
      z = wmma32(qa[1], kb1, z);
      s[sn] = z;
    }
    // online softmax (row = q0 + 8*lh + r lives across the 16 lanes of a half)
    #pragma unroll
    for (int r = 0; r < 8; r++) {
      const int qrow = q0 + lh * 8 + r;
      float s0 = s[0][r] * 0.125f, s1 = s[1][r] * 0.125f;
      if (kbase + lm > qrow)      s0 = -1e30f;
      if (kbase + 16 + lm > qrow) s1 = -1e30f;
      float rm = fmaxf(s0, s1);
      rm = fmaxf(rm, __shfl_xor(rm, 8, 16));
      rm = fmaxf(rm, __shfl_xor(rm, 4, 16));
      rm = fmaxf(rm, __shfl_xor(rm, 2, 16));
      rm = fmaxf(rm, __shfl_xor(rm, 1, 16));
      const float nm   = fmaxf(mi[r], rm);
      const float corr = __expf(mi[r] - nm);
      const float p0 = __expf(s0 - nm), p1 = __expf(s1 - nm);
      float ps = p0 + p1;
      ps += __shfl_xor(ps, 8, 16);
      ps += __shfl_xor(ps, 4, 16);
      ps += __shfl_xor(ps, 2, 16);
      ps += __shfl_xor(ps, 1, 16);
      li[r] = li[r] * corr + ps;
      mi[r] = nm;
      #pragma unroll
      for (int nd = 0; nd < 4; nd++) oacc[nd][r] *= corr;
      pw[(lh * 8 + r) * 32 + lm]      = (f16)p0;
      pw[(lh * 8 + r) * 32 + 16 + lm] = (f16)p1;
    }
    asm volatile("s_wait_dscnt 0" ::: "memory");   // P stores visible (intra-wave)
#if HAS_TDM
    wait_tensor();                                 // V DMA landed (this wave's TENSORcnt)
#else
    wait_async();                                  // V DMA landed (this wave's ASYNCcnt)
#endif
    __builtin_amdgcn_wave_barrier();
    // P as A-frag
    const f16* pp = pw + lm * 32;
    v16h pa = cat16(*(const v8h*)(pp + lh * 16), *(const v8h*)(pp + lh * 16 + 8));
    // O += P * V
    #pragma unroll
    for (int nd = 0; nd < 4; nd++) {
      v16h vb;
      #pragma unroll
      for (int v = 0; v < 8; v++) {
        vb[2 * v]     = vw[(lh * 16 + 2 * v) * 72 + nd * 16 + lm];
        vb[2 * v + 1] = vw[(lh * 16 + 2 * v + 1) * 72 + nd * 16 + lm];
      }
      oacc[nd] = wmma32(pa, vb, oacc[nd]);
    }
    __builtin_amdgcn_wave_barrier();
  }
  #pragma unroll
  for (int nd = 0; nd < 4; nd++)
    #pragma unroll
    for (int r = 0; r < 8; r++) {
      const int row = q0 + lh * 8 + r;
      Y[(rowbase + row) * DD + cb + nd * 16 + lm] = (f16)(oacc[nd][r] / li[r]);
    }
}

// ---------------- host orchestration ----------------
extern "C" void kernel_launch(void* const* d_in, const int* in_sizes, int n_in,
                              void* d_out, int out_size, void* d_ws, size_t ws_size,
                              hipStream_t stream) {
  const int*   idx   = (const int*)  d_in[0];
  const float* tok   = (const float*)d_in[1];
  const float* pos   = (const float*)d_in[2];
  const float* ln1w  = (const float*)d_in[3];
  const float* ln1b  = (const float*)d_in[4];
  const float* Wq    = (const float*)d_in[5];
  const float* bq    = (const float*)d_in[6];
  const float* Wk    = (const float*)d_in[7];
  const float* bk    = (const float*)d_in[8];
  const float* Wv    = (const float*)d_in[9];
  const float* bv    = (const float*)d_in[10];
  const float* Wo    = (const float*)d_in[11];
  const float* bo    = (const float*)d_in[12];
  const float* ln2w  = (const float*)d_in[13];
  const float* ln2b  = (const float*)d_in[14];
  const float* W1    = (const float*)d_in[15];
  const float* b1    = (const float*)d_in[16];
  const float* W2    = (const float*)d_in[17];
  const float* b2    = (const float*)d_in[18];
  const float* lnfw  = (const float*)d_in[19];
  const float* lnfb  = (const float*)d_in[20];
  const float* headw = (const float*)d_in[21];
  float* out = (float*)d_out;

  const size_t R = (size_t)BB * TT;   // 2048 token rows
  char* p = (char*)d_ws;
  auto carve = [&](size_t bytes) { char* r = p; p += (bytes + 255) & ~(size_t)255; return (void*)r; };
  float* xres = (float*)carve(R * DD * 4);
  f16* h16  = (f16*)carve(R * DD * 2);
  f16* q16  = (f16*)carve(R * DD * 2);
  f16* k16  = (f16*)carve(R * DD * 2);
  f16* v16c = (f16*)carve(R * DD * 2);
  f16* y16  = (f16*)carve(R * DD * 2);
  f16* a16  = (f16*)carve(R * 4 * DD * 2);
  f16* wq16 = (f16*)carve((size_t)LL * DD * DD * 2);          // all weights stored [N][K]
  f16* wk16 = (f16*)carve((size_t)LL * DD * DD * 2);
  f16* wv16 = (f16*)carve((size_t)LL * DD * DD * 2);
  f16* wo16 = (f16*)carve((size_t)LL * DD * DD * 2);
  f16* w116 = (f16*)carve((size_t)LL * DD * 4 * DD * 2);
  f16* w216 = (f16*)carve((size_t)LL * DD * 4 * DD * 2);
  f16* hw16 = (f16*)carve((size_t)DD * VV * 2);

  dim3 blk(256);
  embed_k<<<dim3((unsigned)(R * DD / 256)), blk, 0, stream>>>(idx, tok, pos, xres);

  auto cvtT = [&](const float* s, f16* d, int K, int N) {   // [K,N] f32 -> [N,K] f16
    cvtT_k<<<dim3(N / 32, K / 32), dim3(32, 8), 0, stream>>>(s, d, K, N);
  };
  for (int l = 0; l < LL; l++) {
    cvtT(Wq + (size_t)l * DD * DD,     wq16 + (size_t)l * DD * DD,     DD, DD);
    cvtT(Wk + (size_t)l * DD * DD,     wk16 + (size_t)l * DD * DD,     DD, DD);
    cvtT(Wv + (size_t)l * DD * DD,     wv16 + (size_t)l * DD * DD,     DD, DD);
    cvtT(Wo + (size_t)l * DD * DD,     wo16 + (size_t)l * DD * DD,     DD, DD);
    cvtT(W1 + (size_t)l * DD * 4 * DD, w116 + (size_t)l * DD * 4 * DD, DD, 4 * DD);
    cvtT(W2 + (size_t)l * 4 * DD * DD, w216 + (size_t)l * 4 * DD * DD, 4 * DD, DD);
  }
  cvtT(headw, hw16, DD, VV);

  const dim3 gD(DD / 128, (unsigned)(R / 128));       // N=1024 GEMMs
  const dim3 g4D(4 * DD / 128, (unsigned)(R / 128));  // N=4096 GEMM
  for (int l = 0; l < LL; l++) {
    ln_k<<<dim3((unsigned)R), blk, 0, stream>>>(xres, ln1w + l * DD, ln1b + l * DD, h16);
    gemm_k<0><<<gD, blk, 0, stream>>>(h16, wq16 + (size_t)l * DD * DD, bq + l * DD, nullptr, q16,  nullptr, (int)R, DD, DD);
    gemm_k<0><<<gD, blk, 0, stream>>>(h16, wk16 + (size_t)l * DD * DD, bk + l * DD, nullptr, k16,  nullptr, (int)R, DD, DD);
    gemm_k<0><<<gD, blk, 0, stream>>>(h16, wv16 + (size_t)l * DD * DD, bv + l * DD, nullptr, v16c, nullptr, (int)R, DD, DD);
    attn_k<<<dim3(BB * HH * (TT / 128)), blk, 0, stream>>>(q16, k16, v16c, y16);
    gemm_k<1><<<gD, blk, 0, stream>>>(y16, wo16 + (size_t)l * DD * DD, bo + l * DD, xres, nullptr, nullptr, (int)R, DD, DD);
    ln_k<<<dim3((unsigned)R), blk, 0, stream>>>(xres, ln2w + l * DD, ln2b + l * DD, h16);
    gemm_k<2><<<g4D, blk, 0, stream>>>(h16, w116 + (size_t)l * DD * 4 * DD, b1 + l * 4 * DD, nullptr, a16, nullptr, (int)R, 4 * DD, DD);
    gemm_k<1><<<gD, blk, 0, stream>>>(a16, w216 + (size_t)l * 4 * DD * DD, b2 + l * DD, xres, nullptr, nullptr, (int)R, DD, 4 * DD);
  }
  ln_k<<<dim3((unsigned)R), blk, 0, stream>>>(xres, lnfw, lnfb, h16);
  gemm_k<3><<<dim3(VV / 128, (unsigned)(R / 128)), blk, 0, stream>>>(h16, hw16, nullptr, nullptr, nullptr, out, (int)R, VV, DD);
}
